// Conv2d_74139725463813
// MI455X (gfx1250) — compile-verified, matrix operand order checked by reading, not run on hardware
//
#include <hip/hip_runtime.h>

typedef _Float16 half8 __attribute__((ext_vector_type(8)));
typedef _Float16 v16h  __attribute__((ext_vector_type(16)));
typedef float    v8f   __attribute__((ext_vector_type(8)));

// Problem constants (match reference)
#define BN   32
#define CIN  64
#define COUT 64
#define HH   112
#define WW   112
#define KDIM 576          // CIN * 9
#define HSEG 14           // 112 = 8 * 14
#define CIP  72           // ci dim padded 64 -> 72 halves (bank spread + 16B align)
#define LDSW_BYTES (COUT * KDIM * 2)        // 73728 B: all weights as f16, chunk-ordered
#define LDSX_BYTES (3 * 18 * CIP * 2)       // 7776 B: rolling 3-row input cache
#define SHMEM_BYTES (LDSW_BYTES + LDSX_BYTES)

__global__ __launch_bounds__(128) void conv3x3_igemm_wmma(
    const float* __restrict__ x,     // [B, CIN, H, W]
    const float* __restrict__ wgt,   // [COUT, CIN, 3, 3]
    const float* __restrict__ bias,  // [COUT]
    float* __restrict__ out)         // [B, COUT, H, W]
{
    extern __shared__ __align__(16) char smem[];
    _Float16* lds_w = (_Float16*)smem;                 // [COUT][KDIM]
    _Float16* lds_x = (_Float16*)(smem + LDSW_BYTES);  // [3 slot][18 w][CIP ci]

    const int tid   = threadIdx.x;
    const int lane  = tid & 31;
    const int wave  = tid >> 5;          // 0..3 -> co tile
    const int wtile = blockIdx.x;        // 0..6
    const int hseg  = blockIdx.y;        // 0..7
    const int n     = blockIdx.z;        // 0..31
    const int w0      = wtile * 16;
    const int co_base = wave * 16;

    // ---- Stage all weights into LDS as f16, K-chunk ordered (division-free):
    // chunk t = r*2 + ch, r = kh*3+kw, ci = ch*32 + k2
    {
        const int co  = tid >> 1;            // 0..63
        const int kk0 = (tid & 1) * 288;     // each half-thread covers 288 of 576
        for (int j = 0; j < 288; ++j) {
            const int kk = kk0 + j;
            const int t  = kk >> 5;
            const int k2 = kk & 31;
            const int r  = t >> 1;
            const int ci = ((t & 1) << 5) + k2;
            lds_w[co * KDIM + kk] = (_Float16)wgt[(co * CIN + ci) * 9 + r];
        }
    }

    // ---- Per-lane constants (ISA 16-bit A/B fragment striping)
    const int hiHalf  = lane >> 4;
    const int mrow    = lane & 15;
    const int baseSel = hiHalf ? 8 : 0;

    float biasr[8];
#pragma unroll
    for (int r2 = 0; r2 < 8; ++r2)
        biasr[r2] = bias[co_base + (hiHalf << 3) + r2];

    const _Float16* wbase = lds_w + (co_base + mrow) * KDIM;

    // ---- Rolling input-row pipeline over [3][18 w][CIP ci] LDS tiles.
    // fill_load: issue 9 global loads for a row into registers (no waits here).
    // fill_store: convert + ds_store those registers into an LDS slot.
    const int wlc  = (tid & 15) + 1;     // core LDS w index 1..16
    const int ci0  = tid >> 4;           // 0..7
    const int side = tid & 1;            // halo side
    const int hci  = tid >> 1;           // halo ci 0..63
    const int hwc  = side ? (w0 + 16) : (w0 - 1);
    const bool halo_ok = (unsigned)hwc < (unsigned)WW;

    auto fill_load = [&](int row, float (&cv)[8], float& hv) {
        const bool vh = (unsigned)row < (unsigned)HH;   // block-uniform
        if (vh) {
            const float* base =
                x + (((size_t)n * CIN + ci0) * HH + (size_t)row) * WW + w0 + (tid & 15);
#pragma unroll
            for (int k = 0; k < 8; ++k)
                cv[k] = base[(size_t)k * 8 * HH * WW];   // ci += 8 per step
        } else {
#pragma unroll
            for (int k = 0; k < 8; ++k) cv[k] = 0.0f;
        }
        hv = 0.0f;
        if (vh && halo_ok)
            hv = x[(((size_t)n * CIN + hci) * HH + (size_t)row) * WW + hwc];
    };

    auto fill_store = [&](int slot, const float (&cv)[8], float hv) {
        _Float16* dst = lds_x + slot * (18 * CIP);
#pragma unroll
        for (int k = 0; k < 8; ++k)
            dst[wlc * CIP + (ci0 + 8 * k)] = (_Float16)cv[k];
        dst[(side ? 17 : 0) * CIP + hci] = (_Float16)hv;
    };

    // ---- Prologue: rows h0-1 -> slot 0, h0 -> slot 1; h0+1 staged in regs.
    const int h0 = hseg * HSEG;
    int s_m1 = 0, s_0 = 1, s_p1 = 2;     // slots for rows h-1, h, h+1 (rotated)
    {
        float cv[8]; float hv;
        fill_load(h0 - 1, cv, hv);  fill_store(0, cv, hv);
        fill_load(h0,     cv, hv);  fill_store(1, cv, hv);
    }
    float cv[8]; float hv;
    fill_load(h0 + 1, cv, hv);           // in-flight for first iteration

    for (int hh = 0; hh < HSEG; ++hh) {
        const int h = h0 + hh;

        fill_store(s_p1, cv, hv);        // row h+1 -> LDS (waits on its loads)
        __syncthreads();                 // fills visible before compute

        fill_load(h + 2, cv, hv);        // issue next row's loads under compute

        v8f acc = {};
        const int sl[3] = { s_m1, s_0, s_p1 };   // kh = 0,1,2

        for (int kh = 0; kh < 3; ++kh) {
            const _Float16* xslot = lds_x + sl[kh] * (18 * CIP);
            const _Float16* wkh   = wbase + kh * 6 * 32;   // 6 chunks per kh
#pragma unroll
            for (int q = 0; q < 6; ++q) {                  // (kw, ch) pairs
                const int kw = q >> 1;
                const int ch = q & 1;
                const _Float16* brow = xslot + (mrow + kw) * CIP + (ch << 5);
                const _Float16* wrow = wkh + q * 32;
                half8 alo = *(const half8*)(wrow + baseSel);
                half8 ahi = *(const half8*)(wrow + baseSel + 16);
                half8 blo = *(const half8*)(brow + baseSel);
                half8 bhi = *(const half8*)(brow + baseSel + 16);
                v16h a = __builtin_shufflevector(alo, ahi,
                         0,1,2,3,4,5,6,7,8,9,10,11,12,13,14,15);
                v16h b = __builtin_shufflevector(blo, bhi,
                         0,1,2,3,4,5,6,7,8,9,10,11,12,13,14,15);
                acc = __builtin_amdgcn_wmma_f32_16x16x32_f16(
                          false, a, false, b, (short)0, acc, false, false);
            }
        }

        __syncthreads();                 // compute reads done before slot reuse

        // rotate slots: row h+2 will land in the slot that held row h-1
        const int t0 = s_m1; s_m1 = s_0; s_0 = s_p1; s_p1 = t0;

        // C layout: VGPR r -> co = co_base + r + 8*hiHalf, n = mrow
#pragma unroll
        for (int r2 = 0; r2 < 8; ++r2) {
            const int co = co_base + (hiHalf << 3) + r2;
            out[(((size_t)n * COUT + co) * HH + (size_t)h) * WW + w0 + mrow] =
                acc[r2] + biasr[r2];
        }
    }
}

extern "C" void kernel_launch(void* const* d_in, const int* in_sizes, int n_in,
                              void* d_out, int out_size, void* d_ws, size_t ws_size,
                              hipStream_t stream) {
    (void)in_sizes; (void)n_in; (void)out_size; (void)d_ws; (void)ws_size;
    const float* x    = (const float*)d_in[0];
    const float* wgt  = (const float*)d_in[1];
    const float* bias = (const float*)d_in[2];
    float* out        = (float*)d_out;

    (void)hipFuncSetAttribute((const void*)conv3x3_igemm_wmma,
                              hipFuncAttributeMaxDynamicSharedMemorySize,
                              SHMEM_BYTES);

    dim3 grid(WW / 16, 8, BN);   // (7 wtiles, 8 h-segments, 32 batch)
    dim3 block(128);             // 4 waves -> 4 co tiles of 16
    conv3x3_igemm_wmma<<<grid, block, SHMEM_BYTES, stream>>>(x, wgt, bias, out);
}
